// CustomBiLSTM_37323265802558
// MI455X (gfx1250) — compile-verified, matrix operand order checked by reading, not run on hardware
//
#include <hip/hip_runtime.h>

// ---------------------------------------------------------------------------
// BiLSTM (2 layers, B=64, T=1024, D=H=128) for gfx1250.
// Phase A: bf16 WMMA GEMMs for input projections (biases folded), output in
//          WMMA C-fragment order; 4 N-tiles per wave for A-operand reuse,
//          B fragments batch-loaded so WMMAs don't serialize on loadcnt.
// Phase B: persistent-workgroup recurrent scan; Whh fragments pinned in
//          VGPRs, h in LDS (bf16), xg for step t+1 prefetched into cache
//          (global_prefetch_b8) while step t's WMMAs run.
// ---------------------------------------------------------------------------

typedef __attribute__((ext_vector_type(16))) __bf16 bf16x16;
typedef __attribute__((ext_vector_type(8)))  __bf16 bf16x8;
typedef __attribute__((ext_vector_type(8)))  float  f32x8;

static constexpr int Bn = 64;    // batch
static constexpr int Tn = 1024;  // time steps
static constexpr int Hn = 128;   // hidden
static constexpr int Gn = 512;   // 4*H (gates)

#define DEV static __device__ __forceinline__

// Load one 16x32 bf16 A/B fragment (per-lane 16 bf16 = 8 VGPRs).
// p must already include:  row_base + (lane&15)*ldk + (lane>>4)*8
DEV bf16x16 load_frag(const __bf16* p) {
  bf16x8 lo = *(const bf16x8*)p;
  bf16x8 hi = *(const bf16x8*)(p + 16);
  bf16x16 f;
#pragma unroll
  for (int i = 0; i < 8; ++i) { f[i] = lo[i]; f[i + 8] = hi[i]; }
  return f;
}

DEV f32x8 wmma_bf16(bf16x16 a, bf16x16 b, f32x8 c) {
  return __builtin_amdgcn_wmma_f32_16x16x32_bf16(
      /*neg_a=*/false, a, /*neg_b=*/false, b,
      /*c_mod=*/(short)0, c, /*reuse_a=*/false, /*reuse_b=*/false);
}

DEV float sigm(float x)      { return 1.0f / (1.0f + __expf(-x)); }
DEV float tanh_fast(float x) { float e = __expf(-2.0f * x); return (1.0f - e) / (1.0f + e); }

// --------------------------- helper kernels --------------------------------

__global__ void f32_to_bf16_kernel(const float* __restrict__ src,
                                   __bf16* __restrict__ dst, int n) {
  for (int i = blockIdx.x * blockDim.x + threadIdx.x; i < n;
       i += gridDim.x * blockDim.x)
    dst[i] = (__bf16)src[i];
}

// x: [B,T,D] fp32  ->  xbf: [T,B,D] bf16 (time-major for the GEMM)
__global__ void transpose_x_kernel(const float* __restrict__ x,
                                   __bf16* __restrict__ xbf) {
  const int n = Tn * Bn * 128;
  for (int i = blockIdx.x * blockDim.x + threadIdx.x; i < n;
       i += gridDim.x * blockDim.x) {
    int d = i & 127;
    int b = (i >> 7) & 63;
    int t = i >> 13;
    xbf[i] = (__bf16)x[((size_t)b * Tn + t) * 128 + d];
  }
}

__global__ void bias_sum_kernel(const float* __restrict__ bih,
                                const float* __restrict__ bhh,
                                float* __restrict__ bsum) {
  int i = blockIdx.x * blockDim.x + threadIdx.x;
  if (i < Gn) bsum[i] = bih[i] + bhh[i];
}

// --------------------------- gate GEMM -------------------------------------
// xg[M=T*B, N=512] = A[M,K] * W[N,K]^T + bsum.
// One wave per (16-row M-tile) x (4 consecutive 16-col N-tiles): each A
// fragment is loaded once and reused by 4 chained WMMAs (4x less A traffic).
// All 5 fragments of a K-step are loaded before the WMMAs so the loads of
// the next (unrolled) iteration overlap this iteration's WMMAs.
// Output stored in WMMA C-fragment order:
//   xg[ t ][ ct(32) ][ rt(4) ][ lane(32) ][ v(8) ]   (fp32)
__global__ __launch_bounds__(256)
void gates_gemm_kernel(const __bf16* __restrict__ A,
                       const __bf16* __restrict__ Wfw, const __bf16* __restrict__ Wbw,
                       const float* __restrict__ bsfw, const float* __restrict__ bsbw,
                       float* __restrict__ xgfw, float* __restrict__ xgbw,
                       int K) {
  const __bf16* W;  const float* bs;  float* xg;
  if (blockIdx.z == 0) { W = Wfw; bs = bsfw; xg = xgfw; }
  else                 { W = Wbw; bs = bsbw; xg = xgbw; }

  const int lane = threadIdx.x;
  const int wv   = threadIdx.y;
  const int mt   = blockIdx.y * 8 + wv;  // 0..4095 row tiles over M=65536
  const int ct0  = blockIdx.x * 4;       // first of 4 col tiles over N=512
  const int half = lane >> 4, l15 = lane & 15;

  const __bf16* arow = A + (size_t)(mt * 16 + l15) * K + half * 8;
  const __bf16* wrow[4];
#pragma unroll
  for (int c4 = 0; c4 < 4; ++c4)
    wrow[c4] = W + (size_t)((ct0 + c4) * 16 + l15) * K + half * 8;

  f32x8 acc[4];
#pragma unroll
  for (int c4 = 0; c4 < 4; ++c4) acc[c4] = {0.f, 0.f, 0.f, 0.f, 0.f, 0.f, 0.f, 0.f};

#pragma unroll 2
  for (int kk = 0; kk < K; kk += 32) {
    bf16x16 a = load_frag(arow + kk);
    bf16x16 b[4];
#pragma unroll
    for (int c4 = 0; c4 < 4; ++c4) b[c4] = load_frag(wrow[c4] + kk);
#pragma unroll
    for (int c4 = 0; c4 < 4; ++c4) acc[c4] = wmma_bf16(a, b[c4], acc[c4]);
  }

#pragma unroll
  for (int c4 = 0; c4 < 4; ++c4) {
    const float bias = bs[(ct0 + c4) * 16 + l15];
#pragma unroll
    for (int v = 0; v < 8; ++v) acc[c4][v] += bias;
    // fragment-order store: t = mt>>2, rt = mt&3
    size_t off = ((((size_t)(mt >> 2) * 32 + (ct0 + c4)) * 4 + (size_t)(mt & 3)) * 32 + lane) * 8;
    *(f32x8*)(xg + off) = acc[c4];
  }
}

// --------------------------- recurrent scan --------------------------------
// One workgroup per direction (grid.x = 2). 8 waves; wave w owns hidden
// columns [16w, 16w+16) and computes gate tiles i/f/g/o for them.
// Whh fragments live in VGPRs; h lives in LDS as bf16 [64][128].
__global__ __launch_bounds__(256)
void lstm_scan_kernel(const float* __restrict__ xg_fw, const float* __restrict__ xg_bw,
                      const __bf16* __restrict__ Whh_fw, const __bf16* __restrict__ Whh_bw,
                      __bf16* __restrict__ out_bf,   // time-major [T,B,256] or null
                      float*  __restrict__ out_f32)  // batch-major [B,T,256] or null
{
  extern __shared__ __bf16 hsm[];  // [64][128] bf16 hidden state

  const bool rev = (blockIdx.x == 1);
  const float*  xg = rev ? xg_bw : xg_fw;
  const __bf16* W  = rev ? Whh_bw : Whh_fw;
  const int colbase = rev ? Hn : 0;

  const int lane = threadIdx.x & 31;
  const int wv   = threadIdx.x >> 5;   // wave 0..7
  const int half = lane >> 4, l15 = lane & 15;

  // zero h
  {
    bf16x8 z;
#pragma unroll
    for (int i = 0; i < 8; ++i) z[i] = (__bf16)0.0f;
    for (int i = threadIdx.x; i < Bn * Hn / 8; i += 256) ((bf16x8*)hsm)[i] = z;
  }

  // pin Whh B-fragments in VGPRs: Bf[gate][kchunk]
  bf16x16 Bf[4][4];
#pragma unroll
  for (int g = 0; g < 4; ++g) {
    const int ct = g * 8 + wv;
    const __bf16* wrow = W + (size_t)(ct * 16 + l15) * Hn + half * 8;
#pragma unroll
    for (int k4 = 0; k4 < 4; ++k4) Bf[g][k4] = load_frag(wrow + k4 * 32);
  }

  float c[4][8];
#pragma unroll
  for (int rt = 0; rt < 4; ++rt)
#pragma unroll
    for (int v = 0; v < 8; ++v) c[rt][v] = 0.0f;

  __syncthreads();

  for (int tt = 0; tt < Tn; ++tt) {
    const int t = rev ? (Tn - 1 - tt) : tt;

    // init accumulators directly from xg fragments (loads overlap WMMA)
    f32x8 acc[4][4];
    const float* xgt = xg + (size_t)t * (32 * 4 * 32 * 8);
#pragma unroll
    for (int g = 0; g < 4; ++g)
#pragma unroll
      for (int rt = 0; rt < 4; ++rt)
        acc[g][rt] = *(const f32x8*)(xgt + ((((g * 8 + wv) * 4 + rt) * 32 + lane) * 8));

    // prefetch next step's xg fragments into cache (hides HBM latency
    // behind this step's WMMA + nonlinearity work; no VGPR cost)
    if (tt + 1 < Tn) {
      const int tn = rev ? (t - 1) : (t + 1);
      const float* xgn = xg + (size_t)tn * (32 * 4 * 32 * 8);
#pragma unroll
      for (int g = 0; g < 4; ++g)
#pragma unroll
        for (int rt = 0; rt < 4; ++rt)
          __builtin_prefetch(xgn + ((((g * 8 + wv) * 4 + rt) * 32 + lane) * 8), 0, 3);
    }

    // gates += h @ Whh^T
#pragma unroll
    for (int k4 = 0; k4 < 4; ++k4) {
      bf16x16 Af[4];
#pragma unroll
      for (int rt = 0; rt < 4; ++rt)
        Af[rt] = load_frag(hsm + (rt * 16 + l15) * Hn + k4 * 32 + half * 8);
#pragma unroll
      for (int rt = 0; rt < 4; ++rt)
#pragma unroll
        for (int g = 0; g < 4; ++g)
          acc[g][rt] = wmma_bf16(Af[rt], Bf[g][k4], acc[g][rt]);
    }

    __syncthreads();  // all reads of h(t-1) complete before overwrite

    const int j = 16 * wv + l15;
#pragma unroll
    for (int rt = 0; rt < 4; ++rt) {
#pragma unroll
      for (int v = 0; v < 8; ++v) {
        const float iv = sigm(acc[0][rt][v]);
        const float fv = sigm(acc[1][rt][v]);
        const float gv = tanh_fast(acc[2][rt][v]);
        const float ov = sigm(acc[3][rt][v]);
        const float cc = fv * c[rt][v] + iv * gv;
        c[rt][v] = cc;
        const float hv = ov * tanh_fast(cc);
        const int b = rt * 16 + v + half * 8;   // C-layout: M = v + 8*(lane>>4)
        hsm[b * Hn + j] = (__bf16)hv;
        if (out_bf)  out_bf [((size_t)t * Bn + b) * 256 + colbase + j] = (__bf16)hv;
        if (out_f32) out_f32[((size_t)b * Tn + t) * 256 + colbase + j] = hv;
      }
    }
    __syncthreads();  // new h visible to all waves
  }
}

// --------------------------- launcher --------------------------------------

extern "C" void kernel_launch(void* const* d_in, const int* in_sizes, int n_in,
                              void* d_out, int out_size, void* d_ws, size_t ws_size,
                              hipStream_t stream) {
  (void)in_sizes; (void)n_in; (void)out_size; (void)ws_size;

  const float* x        = (const float*)d_in[0];
  // d_in[1] = lengths (unused by the reference forward)
  const float* Wih_fw1  = (const float*)d_in[2];
  const float* Whh_fw1  = (const float*)d_in[3];
  const float* bih_fw1  = (const float*)d_in[4];
  const float* bhh_fw1  = (const float*)d_in[5];
  const float* Wih_bw1  = (const float*)d_in[6];
  const float* Whh_bw1  = (const float*)d_in[7];
  const float* bih_bw1  = (const float*)d_in[8];
  const float* bhh_bw1  = (const float*)d_in[9];
  const float* Wih_fw2  = (const float*)d_in[10];
  const float* Whh_fw2  = (const float*)d_in[11];
  const float* bih_fw2  = (const float*)d_in[12];
  const float* bhh_fw2  = (const float*)d_in[13];
  const float* Wih_bw2  = (const float*)d_in[14];
  const float* Whh_bw2  = (const float*)d_in[15];
  const float* bih_bw2  = (const float*)d_in[16];
  const float* bhh_bw2  = (const float*)d_in[17];

  char* ws = (char*)d_ws;
  size_t off = 0;
  auto take = [&](size_t bytes) -> char* {
    char* p = ws + off;
    off = (off + bytes + 255) & ~(size_t)255;
    return p;
  };

  __bf16* xbf     = (__bf16*)take((size_t)Tn * Bn * 128 * 2);   // 16 MB
  __bf16* out1bf  = (__bf16*)take((size_t)Tn * Bn * 256 * 2);   // 32 MB
  float*  xg_fw   = (float*) take((size_t)Tn * Bn * Gn * 4);    // 128 MB (reused L1/L2)
  float*  xg_bw   = (float*) take((size_t)Tn * Bn * Gn * 4);    // 128 MB
  __bf16* WihFw1b = (__bf16*)take((size_t)Gn * 128 * 2);
  __bf16* WihBw1b = (__bf16*)take((size_t)Gn * 128 * 2);
  __bf16* WihFw2b = (__bf16*)take((size_t)Gn * 256 * 2);
  __bf16* WihBw2b = (__bf16*)take((size_t)Gn * 256 * 2);
  __bf16* WhhFw1b = (__bf16*)take((size_t)Gn * Hn * 2);
  __bf16* WhhBw1b = (__bf16*)take((size_t)Gn * Hn * 2);
  __bf16* WhhFw2b = (__bf16*)take((size_t)Gn * Hn * 2);
  __bf16* WhhBw2b = (__bf16*)take((size_t)Gn * Hn * 2);
  float*  bsFw1   = (float*) take(Gn * 4);
  float*  bsBw1   = (float*) take(Gn * 4);
  float*  bsFw2   = (float*) take(Gn * 4);
  float*  bsBw2   = (float*) take(Gn * 4);

  // --- prep: transpose/convert x, convert weights, fold biases ---
  transpose_x_kernel<<<4096, 256, 0, stream>>>(x, xbf);
  f32_to_bf16_kernel<<<256, 256, 0, stream>>>(Wih_fw1, WihFw1b, Gn * 128);
  f32_to_bf16_kernel<<<256, 256, 0, stream>>>(Wih_bw1, WihBw1b, Gn * 128);
  f32_to_bf16_kernel<<<512, 256, 0, stream>>>(Wih_fw2, WihFw2b, Gn * 256);
  f32_to_bf16_kernel<<<512, 256, 0, stream>>>(Wih_bw2, WihBw2b, Gn * 256);
  f32_to_bf16_kernel<<<256, 256, 0, stream>>>(Whh_fw1, WhhFw1b, Gn * Hn);
  f32_to_bf16_kernel<<<256, 256, 0, stream>>>(Whh_bw1, WhhBw1b, Gn * Hn);
  f32_to_bf16_kernel<<<256, 256, 0, stream>>>(Whh_fw2, WhhFw2b, Gn * Hn);
  f32_to_bf16_kernel<<<256, 256, 0, stream>>>(Whh_bw2, WhhBw2b, Gn * Hn);
  bias_sum_kernel<<<2, 256, 0, stream>>>(bih_fw1, bhh_fw1, bsFw1);
  bias_sum_kernel<<<2, 256, 0, stream>>>(bih_bw1, bhh_bw1, bsBw1);
  bias_sum_kernel<<<2, 256, 0, stream>>>(bih_fw2, bhh_fw2, bsFw2);
  bias_sum_kernel<<<2, 256, 0, stream>>>(bih_bw2, bhh_bw2, bsBw2);

  const dim3 gblk(32, 8, 1);
  const dim3 ggrd(8, 512, 2);  // 8 groups of 4 N-tiles x 4096 M-tiles x 2 dirs
  const size_t scan_lds = (size_t)Bn * Hn * 2;  // 16 KB

  // --- layer 1 ---
  gates_gemm_kernel<<<ggrd, gblk, 0, stream>>>(xbf, WihFw1b, WihBw1b,
                                               bsFw1, bsBw1, xg_fw, xg_bw, 128);
  lstm_scan_kernel<<<2, 256, scan_lds, stream>>>(xg_fw, xg_bw, WhhFw1b, WhhBw1b,
                                                 out1bf, nullptr);

  // --- layer 2 (xg buffers reused) ---
  gates_gemm_kernel<<<ggrd, gblk, 0, stream>>>(out1bf, WihFw2b, WihBw2b,
                                               bsFw2, bsBw2, xg_fw, xg_bw, 256);
  lstm_scan_kernel<<<2, 256, scan_lds, stream>>>(xg_fw, xg_bw, WhhFw2b, WhhBw2b,
                                                 nullptr, (float*)d_out);
}
